// VLLMDualMLPAdapter_75694503624730
// MI455X (gfx1250) — matrix operand order repaired and checked by reading, not compiled, not run on hardware
//
#include <hip/hip_runtime.h>
#include <math.h>
#include <stdint.h>

// Problem dims (static, match reference)
#define T_DIM 4096
#define H_DIM 4096
#define I_DIM 11008
#define S_N   8
#define RN    64
#define FN    64

// GEMM tiling
#define BM     128
#define BK     64
#define LDS_K  72          // BK + 8 pad (bf16 elements) to spread LDS banks

typedef __bf16 bf16_t;
typedef __attribute__((ext_vector_type(8)))  __bf16 v8bf;
typedef __attribute__((ext_vector_type(16))) __bf16 v16bf;
typedef __attribute__((ext_vector_type(8)))  float  v8f;
typedef __attribute__((ext_vector_type(4)))  int    v4i;

// ---------------------------------------------------------------------------
// CDNA5 async global->LDS copy (ASYNCcnt-tracked), with safe fallback.
// Builtin signature (probe round 2): (int4 AS1*, int4 AS3*, imm int, imm int)
// ---------------------------------------------------------------------------
#if defined(__has_builtin)
#  if __has_builtin(__builtin_amdgcn_global_load_async_to_lds_b128)
#    define ASYNC_LDS 1
#  endif
#endif
#ifndef ASYNC_LDS
#  define ASYNC_LDS 0
#endif

typedef __attribute__((address_space(1))) v4i as1_v4i;
typedef __attribute__((address_space(3))) v4i as3_v4i;

__device__ __forceinline__ void copy16_g2l(const bf16_t* __restrict__ g,
                                           bf16_t* l) {
#if ASYNC_LDS
    __builtin_amdgcn_global_load_async_to_lds_b128(
        (as1_v4i*)(uintptr_t)g,
        (as3_v4i*)(uint32_t)(uintptr_t)l,   // LDS offset = low 32 bits
        0, 0);
#else
    *(v8bf*)l = *(const v8bf*)g;             // sync: global_load + ds_store
#endif
}

__device__ __forceinline__ void wait_async_copies() {
#if ASYNC_LDS
#  if __has_builtin(__builtin_amdgcn_s_wait_asynccnt)
    __builtin_amdgcn_s_wait_asynccnt(0);
#  else
    asm volatile("s_wait_asynccnt 0x0" ::: "memory");
#  endif
#endif
}

// ---------------------------------------------------------------------------
// fp32 -> bf16 conversion (bandwidth-bound pre-pass)
// ---------------------------------------------------------------------------
__global__ __launch_bounds__(256)
void cvt_f32_to_bf16(const float* __restrict__ in, bf16_t* __restrict__ out,
                     size_t n) {
    size_t i = (size_t)blockIdx.x * blockDim.x + threadIdx.x;
    size_t stride = (size_t)gridDim.x * blockDim.x;
    for (; i < n; i += stride) out[i] = (bf16_t)in[i];
}

// ---------------------------------------------------------------------------
// WMMA fragment loads from LDS (wave32 layouts, CDNA5 ISA 7.12.2)
// A 16x32 bf16: lane l: row l&15; K = kb..kb+7 and kb+16..kb+23, kb=(l<16?0:8)
// B 32x16 bf16: lane l: col l&15; K = kb..kb+7 and kb+16..kb+23, kb=(l<16?0:8)
// ---------------------------------------------------------------------------
__device__ __forceinline__ v16bf fragA_lds(const bf16_t* p) {
    v8bf lo = *(const v8bf*)p;
    v8bf hi = *(const v8bf*)(p + 16);
    v16bf a;
#pragma unroll
    for (int i = 0; i < 8; ++i) { a[i] = lo[i]; a[i + 8] = hi[i]; }
    return a;
}
__device__ __forceinline__ v16bf fragB_lds(const bf16_t* p) {
    v8bf lo = *(const v8bf*)p;
    v8bf hi = *(const v8bf*)(p + 16);
    v16bf b;
#pragma unroll
    for (int i = 0; i < 8; ++i) { b[i] = lo[i]; b[i + 8] = hi[i]; }
    return b;
}

__device__ __forceinline__ float silu_mul(float g, float u) {
    return (g / (1.0f + __expf(-g))) * u;
}

// ---------------------------------------------------------------------------
// GEMM1 + fused SwiGLU: act = silu(x@Wg.T) * (x@Wu.T)  -> bf16 [T, I]
// Block tile: 128 M x 64 N(I), staging both gate rows (LDS B rows 0..63)
// and up rows (64..127). 8 waves = 2(M) x 4(N); each wave: 4 M-subtiles,
// one N tile, dual accumulators (gate/up). 16 WMMAs per 64-K stage.
// ---------------------------------------------------------------------------
__global__ __launch_bounds__(256)
void gemm1_swiglu(const bf16_t* __restrict__ xb,
                  const bf16_t* __restrict__ wgu,
                  bf16_t* __restrict__ act) {
    __shared__ bf16_t sA[2][BM * LDS_K];
    __shared__ bf16_t sB[2][128 * LDS_K];

    const int lane = threadIdx.x & 31;
    const int wave = threadIdx.x >> 5;
    const int wm = wave >> 2;                  // 0..1
    const int wn = wave & 3;                   // 0..3
    const int m0 = blockIdx.x * BM;
    const int n0 = blockIdx.y * 64;            // I-space

    const int lm  = lane & 15;
    const int akb = (lane >> 4) << 3;          // 0 / 8
    const int bkb = (lane >> 4) << 3;          // 0 / 8

    v8f accG[4] = {{0}, {0}, {0}, {0}};
    v8f accU[4] = {{0}, {0}, {0}, {0}};

    auto load_stage = [&](int kt, int buf) {
        const int k0 = kt * BK;
#pragma unroll
        for (int c = 0; c < 4; ++c) {
            const int chunk = (int)threadIdx.x + c * 256;   // 1024 chunks of 16B
            const int row = chunk >> 3;
            const int col = (chunk & 7) << 3;
            copy16_g2l(xb + (size_t)(m0 + row) * H_DIM + k0 + col,
                       &sA[buf][row * LDS_K + col]);
        }
#pragma unroll
        for (int c = 0; c < 4; ++c) {
            const int chunk = (int)threadIdx.x + c * 256;
            const int r = chunk >> 3;
            const int col = (chunk & 7) << 3;
            const size_t grow = (r < 64) ? (size_t)(n0 + r)
                                         : (size_t)(I_DIM + n0 + (r - 64));
            copy16_g2l(wgu + grow * H_DIM + k0 + col,
                       &sB[buf][r * LDS_K + col]);
        }
    };

    load_stage(0, 0);
    wait_async_copies();
    __syncthreads();

    const int NT = H_DIM / BK;                 // 64 stages
    for (int kt = 0; kt < NT; ++kt) {
        const int cur = kt & 1;
        if (kt + 1 < NT) load_stage(kt + 1, cur ^ 1);

#pragma unroll
        for (int ks = 0; ks < BK; ks += 32) {
            v16bf a[4];
#pragma unroll
            for (int i = 0; i < 4; ++i)
                a[i] = fragA_lds(&sA[cur][(wm * 64 + i * 16 + lm) * LDS_K + ks + akb]);
            v16bf bg = fragB_lds(&sB[cur][(wn * 16 + lm) * LDS_K + ks + bkb]);
            v16bf bu = fragB_lds(&sB[cur][(64 + wn * 16 + lm) * LDS_K + ks + bkb]);
#pragma unroll
            for (int i = 0; i < 4; ++i) {
                accG[i] = __builtin_amdgcn_wmma_f32_16x16x32_bf16(
                    false, a[i], false, bg, (short)0, accG[i], false, false);
                accU[i] = __builtin_amdgcn_wmma_f32_16x16x32_bf16(
                    false, a[i], false, bu, (short)0, accU[i], false, false);
            }
        }
        wait_async_copies();
        __syncthreads();
    }

    // SwiGLU epilogue in registers -> bf16 act
    const int r0 = m0 + wm * 64 + ((lane >> 4) << 3);
    const int c0 = n0 + wn * 16 + lm;
#pragma unroll
    for (int i = 0; i < 4; ++i)
#pragma unroll
        for (int v = 0; v < 8; ++v)
            act[(size_t)(r0 + i * 16 + v) * I_DIM + c0] =
                (bf16_t)silu_mul(accG[i][v], accU[i][v]);
}

// ---------------------------------------------------------------------------
// GEMM2: out = act @ down_w.T  (f32 out [T, H])
// Block tile 128 M x 128 N(H) x 64 K; 8 waves = 2(M) x 4(N);
// each wave: 4 M-subtiles x 2 N-subtiles = 8 accumulators, 16 WMMAs/stage.
// ---------------------------------------------------------------------------
__global__ __launch_bounds__(256)
void gemm2_down(const bf16_t* __restrict__ act,
                const bf16_t* __restrict__ wd,
                float* __restrict__ out) {
    __shared__ bf16_t sA[2][BM * LDS_K];
    __shared__ bf16_t sB[2][128 * LDS_K];

    const int lane = threadIdx.x & 31;
    const int wave = threadIdx.x >> 5;
    const int wm = wave >> 2;
    const int wn = wave & 3;
    const int m0 = blockIdx.x * BM;
    const int n0 = blockIdx.y * 128;           // H-space

    const int lm  = lane & 15;
    const int akb = (lane >> 4) << 3;
    const int bkb = (lane >> 4) << 3;

    v8f acc[4][2] = {{{0}, {0}}, {{0}, {0}}, {{0}, {0}}, {{0}, {0}}};

    auto load_stage = [&](int kt, int buf) {
        const int k0 = kt * BK;
#pragma unroll
        for (int c = 0; c < 4; ++c) {
            const int chunk = (int)threadIdx.x + c * 256;
            const int row = chunk >> 3;
            const int col = (chunk & 7) << 3;
            copy16_g2l(act + (size_t)(m0 + row) * I_DIM + k0 + col,
                       &sA[buf][row * LDS_K + col]);
        }
#pragma unroll
        for (int c = 0; c < 4; ++c) {
            const int chunk = (int)threadIdx.x + c * 256;
            const int r = chunk >> 3;
            const int col = (chunk & 7) << 3;
            copy16_g2l(wd + (size_t)(n0 + r) * I_DIM + k0 + col,
                       &sB[buf][r * LDS_K + col]);
        }
    };

    load_stage(0, 0);
    wait_async_copies();
    __syncthreads();

    const int NT = I_DIM / BK;                 // 172 stages
    for (int kt = 0; kt < NT; ++kt) {
        const int cur = kt & 1;
        if (kt + 1 < NT) load_stage(kt + 1, cur ^ 1);

#pragma unroll
        for (int ks = 0; ks < BK; ks += 32) {
            v16bf a[4];
#pragma unroll
            for (int i = 0; i < 4; ++i)
                a[i] = fragA_lds(&sA[cur][(wm * 64 + i * 16 + lm) * LDS_K + ks + akb]);
            v16bf b0 = fragB_lds(&sB[cur][(wn * 32 + lm) * LDS_K + ks + bkb]);
            v16bf b1 = fragB_lds(&sB[cur][(wn * 32 + 16 + lm) * LDS_K + ks + bkb]);
#pragma unroll
            for (int i = 0; i < 4; ++i) {
                acc[i][0] = __builtin_amdgcn_wmma_f32_16x16x32_bf16(
                    false, a[i], false, b0, (short)0, acc[i][0], false, false);
                acc[i][1] = __builtin_amdgcn_wmma_f32_16x16x32_bf16(
                    false, a[i], false, b1, (short)0, acc[i][1], false, false);
            }
        }
        wait_async_copies();
        __syncthreads();
    }

    const int r0 = m0 + wm * 64 + ((lane >> 4) << 3);
    const int c0 = n0 + wn * 32 + lm;
#pragma unroll
    for (int i = 0; i < 4; ++i)
#pragma unroll
        for (int j = 0; j < 2; ++j)
#pragma unroll
            for (int v = 0; v < 8; ++v)
                out[(size_t)(r0 + i * 16 + v) * H_DIM + c0 + j * 16] = acc[i][j][v];
}

// ---------------------------------------------------------------------------
// Token-routed dual SwiGLU adapters (<3% of FLOPs): VALU, one block per token.
// ---------------------------------------------------------------------------
__global__ __launch_bounds__(256)
void adapters_routed(const float* __restrict__ x,
                     const float* __restrict__ rg, const float* __restrict__ ru,
                     const float* __restrict__ rd,
                     const float* __restrict__ fg, const float* __restrict__ fu,
                     const float* __restrict__ fd,
                     const float* __restrict__ scales,
                     const int* __restrict__ tok,
                     float* __restrict__ out) {
    __shared__ float inter[RN + FN];

    const int t = blockIdx.x;
    const int s = tok[t];
    const int lane = threadIdx.x & 31;
    const int wave = threadIdx.x >> 5;
    const float* xt = x + (size_t)t * H_DIM;

#pragma unroll 1
    for (int i = 0; i < 16; ++i) {
        const int j = wave * 16 + i;  // 0..127
        const float* gw;
        const float* uw;
        if (j < RN) {
            gw = rg + ((size_t)s * RN + j) * H_DIM;
            uw = ru + ((size_t)s * RN + j) * H_DIM;
        } else {
            gw = fg + ((size_t)s * FN + (j - RN)) * H_DIM;
            uw = fu + ((size_t)s * FN + (j - RN)) * H_DIM;
        }
        float dg = 0.0f, du = 0.0f;
        for (int k = lane; k < H_DIM; k += 32) {
            const float xv = xt[k];
            dg = fmaf(xv, gw[k], dg);
            du = fmaf(xv, uw[k], du);
        }
#pragma unroll
        for (int off = 16; off > 0; off >>= 1) {
            dg += __shfl_xor(dg, off, 32);
            du += __shfl_xor(du, off, 32);
        }
        if (lane == 0) inter[j] = silu_mul(dg, du);
    }
    __syncthreads();

    const float s0 = scales[s * 2 + 0];
    const float s1 = scales[s * 2 + 1];
    for (int h = threadIdx.x; h < H_DIM; h += 256) {
        const float* rdw = rd + ((size_t)s * H_DIM + h) * RN;
        const float* fdw = fd + ((size_t)s * H_DIM + h) * FN;
        float accR = 0.0f, accF = 0.0f;
#pragma unroll 4
        for (int n = 0; n < RN; ++n) accR = fmaf(inter[n],      rdw[n], accR);
#pragma unroll 4
        for (int n = 0; n < FN; ++n) accF = fmaf(inter[RN + n], fdw[n], accF);
        out[(size_t)t * H_DIM + h] += accR * s0 + accF * s1;
    }
}

// ---------------------------------------------------------------------------
// Launch
// ---------------------------------------------------------------------------
extern "C" void kernel_launch(void* const* d_in, const int* in_sizes, int n_in,
                              void* d_out, int out_size, void* d_ws, size_t ws_size,
                              hipStream_t stream) {
    (void)in_sizes; (void)n_in; (void)out_size; (void)ws_size;

    const float* x   = (const float*)d_in[0];
    const float* wgu = (const float*)d_in[1];
    const float* wd  = (const float*)d_in[2];
    const float* rg  = (const float*)d_in[3];
    const float* ru  = (const float*)d_in[4];
    const float* rd  = (const float*)d_in[5];
    const float* fg  = (const float*)d_in[6];
    const float* fu  = (const float*)d_in[7];
    const float* fd  = (const float*)d_in[8];
    const float* sc  = (const float*)d_in[9];
    const int*   tok = (const int*)d_in[10];
    float* out = (float*)d_out;

    // Workspace layout (bf16): x | gate_up_w | down_w | act
    bf16_t* xb   = (bf16_t*)d_ws;
    bf16_t* wgub = xb   + (size_t)T_DIM * H_DIM;
    bf16_t* wdb  = wgub + (size_t)2 * I_DIM * H_DIM;
    bf16_t* actb = wdb  + (size_t)H_DIM * I_DIM;

    cvt_f32_to_bf16<<<4096, 256, 0, stream>>>(x,   xb,   (size_t)T_DIM * H_DIM);
    cvt_f32_to_bf16<<<8192, 256, 0, stream>>>(wgu, wgub, (size_t)2 * I_DIM * H_DIM);
    cvt_f32_to_bf16<<<8192, 256, 0, stream>>>(wd,  wdb,  (size_t)H_DIM * I_DIM);

    dim3 g1(T_DIM / BM, I_DIM / 64);   // 32 x 172
    gemm1_swiglu<<<g1, 256, 0, stream>>>(xb, wgub, actb);

    dim3 g2(T_DIM / BM, H_DIM / 128);  // 32 x 32
    gemm2_down<<<g2, 256, 0, stream>>>(actb, wdb, out);

    adapters_routed<<<T_DIM, 256, 0, stream>>>(x, rg, ru, rd, fg, fu, fd,
                                               sc, tok, out);
}
